// TransAtt_23613730193707
// MI455X (gfx1250) — matrix operand order repaired and checked by reading, not compiled, not run on hardware
//
#include <hip/hip_runtime.h>
#include <hip/hip_bf16.h>
#include <math.h>
#include <stdint.h>

typedef __attribute__((ext_vector_type(16))) _Float16 v16h;
typedef __attribute__((ext_vector_type(8)))  _Float16 v8h;
typedef __attribute__((ext_vector_type(8)))  float    v8f;

#define DEV static __device__ __forceinline__

constexpr int Bn   = 8;
constexpr int H    = 128;
constexpr int Wd   = 128;
constexpr int HW   = H * Wd;          // 16384
constexpr int C_SX1 = 256;
constexpr int C_T   = 64;
constexpr int C1    = C_SX1 + C_T;    // 320  (t1 input)
constexpr int E1    = 128;
constexpr int E2    = 64;
constexpr int C_SX2 = 256;
constexpr int EBD   = 128;
constexpr int CB    = EBD + C_T;      // 192  (bcat channels)
constexpr int A1C   = 128;
constexpr int OTC   = 256;
constexpr int KOT   = CB * 9;         // 1728 (3x3 conv, K = tap*192 + c)

// LDS pitches (halves): 16B aligned, pitch/2 dwords spreads 16B lane
// accesses across the 64 banks (~2-way worst case).
constexpr int P320 = 328;
constexpr int P256 = 264;
constexpr int P192 = 200;
constexpr int P128 = 136;

// ---------------------------------------------------------------------------
DEV v8f wmma_step(v16h a, v16h b, v8f c) {
  return __builtin_amdgcn_wmma_f32_16x16x32_f16(
      false, a, false, b, (short)0, c, false, false);
}

// A fragment (16x32) from row-major f16 matrix; both 8-half chunks 16B aligned.
DEV v16h load_a_f16(const _Float16* __restrict__ A, int lda, int m0, int k0,
                    int lane) {
  const int lr = lane & 15;
  const int kb = k0 + ((lane & 16) ? 16 : 0);
  const _Float16* p = A + (size_t)(m0 + lr) * lda + kb;
  v8h lo = *(const v8h*)(p);
  v8h hi = *(const v8h*)(p + 8);
  return __builtin_shufflevector(lo, hi, 0, 1, 2, 3, 4, 5, 6, 7,
                                 8, 9, 10, 11, 12, 13, 14, 15);
}

// B fragment (32x16) from transposed LDS tile ([col][k]): 16 contiguous halves.
DEV v16h load_b_ldsT(const _Float16* colbase, int k0, int lane) {
  const int kb = k0 + ((lane & 16) ? 16 : 0);
  v8h lo = *(const v8h*)(colbase + kb);
  v8h hi = *(const v8h*)(colbase + kb + 8);
  return __builtin_shufflevector(lo, hi, 0, 1, 2, 3, 4, 5, 6, 7,
                                 8, 9, 10, 11, 12, 13, 14, 15);
}

// CDNA5 async global->LDS copy (16B per lane, tracked by ASYNCcnt).
DEV void async_load_b128(uint32_t lds_off, const _Float16* g) {
  asm volatile("global_load_async_to_lds_b128 %0, %1, off"
               :: "v"(lds_off), "v"((uint64_t)(uintptr_t)g)
               : "memory");
}
DEV void wait_async0() { asm volatile("s_wait_asynccnt 0x0" ::: "memory"); }
DEV uint32_t lds_off(const void* p) { return (uint32_t)(uintptr_t)p; }

// ---------------------------------------------------------------------------
// Prep kernels
// ---------------------------------------------------------------------------
__global__ void kcvt_f16(const float* __restrict__ src,
                         _Float16* __restrict__ dst, int n) {
  int i = blockIdx.x * blockDim.x + threadIdx.x;
  if (i < n) dst[i] = (_Float16)src[i];
}

// OIHW [256][192][3][3] -> [o][tap][c] f16 so K = tap*192 + c is row-major.
__global__ void kperm_ot(const float* __restrict__ src,
                         _Float16* __restrict__ dst) {
  int i = blockIdx.x * blockDim.x + threadIdx.x;
  if (i >= OTC * KOT) return;
  const int o = i / KOT, r = i % KOT;
  const int tap = r / CB, c = r % CB;
  dst[i] = (_Float16)src[(size_t)o * KOT + c * 9 + tap];
}

__global__ void k0_zero(float* p, int n) {
  int i = blockIdx.x * blockDim.x + threadIdx.x;
  if (i < n) p[i] = 0.f;
}

// ---------------------------------------------------------------------------
// K1: t1 conv1x1 (K=320) + ReLU + GAP partial sums. 64 px per block.
// ---------------------------------------------------------------------------
__global__ __launch_bounds__(256) void k1_t1_gap(
    const float* __restrict__ sx1, const float* __restrict__ tx,
    const _Float16* __restrict__ t1w, const float* __restrict__ t1b,
    float* __restrict__ gsum) {
  __shared__ _Float16 xcat[64 * P320];  // [col][k] 41 KB
  const int b    = blockIdx.y;
  const int p0   = blockIdx.x * 64;
  const int wave = threadIdx.x >> 5;
  const int lane = threadIdx.x & 31;
  const int lr   = lane & 15;
  const bool hi  = (lane & 16) != 0;

  const float* xs1 = sx1 + (size_t)b * C_SX1 * HW;
  const float* xt  = tx  + (size_t)b * C_T   * HW;
  for (int i = threadIdx.x; i < C1 * 64; i += 256) {
    const int c = i >> 6, p = i & 63;   // p fastest: coalesced global
    const float v = (c < C_SX1) ? xs1[(size_t)c * HW + p0 + p]
                                : xt[(size_t)(c - C_SX1) * HW + p0 + p];
    xcat[p * P320 + c] = (_Float16)v;
  }
  __syncthreads();

  const int mt = wave;
  v8f acc[4] = {};
  for (int ks = 0; ks < C1 / 32; ++ks) {
    const int k0 = ks * 32;
    v16h a = load_a_f16(t1w, C1, mt * 16, k0, lane);
#pragma unroll
    for (int nt = 0; nt < 4; ++nt) {
      v16h bf = load_b_ldsT(&xcat[(nt * 16 + lr) * P320], k0, lane);
      acc[nt] = wmma_step(a, bf, acc[nt]);
    }
  }

  float rsum[8];
#pragma unroll
  for (int r = 0; r < 8; ++r) rsum[r] = 0.f;
#pragma unroll
  for (int nt = 0; nt < 4; ++nt)
#pragma unroll
    for (int r = 0; r < 8; ++r) {
      const int row = mt * 16 + (hi ? 8 : 0) + r;
      rsum[r] += fmaxf(acc[nt][r] + t1b[row], 0.f);
    }
#pragma unroll
  for (int r = 0; r < 8; ++r) {
#pragma unroll
    for (int off = 1; off < 16; off <<= 1)
      rsum[r] += __shfl_xor(rsum[r], off, 32);
  }
  if (lr == 0) {
#pragma unroll
    for (int r = 0; r < 8; ++r)
      atomicAdd(&gsum[b * E1 + mt * 16 + (hi ? 8 : 0) + r], rsum[r]);
  }
}

// ---------------------------------------------------------------------------
// K2: per-sample transfer MLP + L2 normalize (tiny; plain VALU)
// ---------------------------------------------------------------------------
__global__ __launch_bounds__(256) void k2_transfer(
    const float* __restrict__ gsum,
    const float* __restrict__ t2w, const float* __restrict__ t3w,
    float* __restrict__ wvec, float* __restrict__ wout) {
  __shared__ float g[E1];
  __shared__ float hb[E2];
  __shared__ float w3[C_SX2];
  __shared__ float red[256];
  const int b = blockIdx.x;
  const int t = threadIdx.x;

  if (t < E1) g[t] = gsum[b * E1 + t] * (1.f / (float)HW);
  __syncthreads();
  if (t < E2) {
    float s = 0.f;
    for (int c = 0; c < E1; ++c) s += t2w[t * E1 + c] * g[c];
    hb[t] = fmaxf(s, 0.f);
  }
  __syncthreads();
  float w3v = 0.f;
  if (t < C_SX2) {
    float s = 0.f;
    for (int e = 0; e < E2; ++e) s += t3w[t * E2 + e] * hb[e];
    w3[t] = s;
    w3v = s;
  }
  red[t] = w3v * w3v;
  __syncthreads();
  for (int s = 128; s > 0; s >>= 1) {
    if (t < s) red[t] += red[t + s];
    __syncthreads();
  }
  const float nrm = fmaxf(sqrtf(red[0]), 1e-12f);
  if (t < C_SX2) {
    const float v = w3[t] / nrm;
    wvec[b * C_SX2 + t] = v;
    wout[b * C_SX2 + t] = v;
  }
}

// ---------------------------------------------------------------------------
// K3: stage sx2 -> LDS, U-GEMM -> ReLU -> scale -> LDS -> V-GEMM -> ReLU ->
//     bcat (pixel-major f16 [b][p][c]); tx appended as channels 128..191.
// ---------------------------------------------------------------------------
__global__ __launch_bounds__(256) void k3_mod(
    const float* __restrict__ sx2, const float* __restrict__ tx,
    const _Float16* __restrict__ Uw, const float* __restrict__ Ub,
    const _Float16* __restrict__ Vw, const float* __restrict__ Vb,
    const float* __restrict__ wvec, _Float16* __restrict__ bcat) {
  __shared__ _Float16 xs[64 * P256];  // [col][k] sx2 tile, 33 KB
  __shared__ _Float16 us[64 * P256];  // [col][k] modulated U tile, 33 KB
  const int b    = blockIdx.y;
  const int p0   = blockIdx.x * 64;
  const int wave = threadIdx.x >> 5;
  const int lane = threadIdx.x & 31;
  const int lr   = lane & 15;
  const bool hi  = (lane & 16) != 0;
  const float* x = sx2 + (size_t)b * C_SX2 * HW;
  _Float16* bcp  = bcat + (size_t)b * HW * CB;

  for (int i = threadIdx.x; i < C_SX2 * 64; i += 256) {
    const int c = i >> 6, p = i & 63;
    xs[p * P256 + c] = (_Float16)x[(size_t)c * HW + p0 + p];
  }
  __syncthreads();

  // U (256x256): 16 M-tiles / 8 waves, 4 N-tiles accumulated per A load
  for (int hf = 0; hf < 2; ++hf) {
    const int mt = wave + hf * 8;
    v8f acc[4] = {};
    for (int ks = 0; ks < C_SX2 / 32; ++ks) {
      const int k0 = ks * 32;
      v16h a = load_a_f16(Uw, C_SX2, mt * 16, k0, lane);
#pragma unroll
      for (int nt = 0; nt < 4; ++nt) {
        v16h bf = load_b_ldsT(&xs[(nt * 16 + lr) * P256], k0, lane);
        acc[nt] = wmma_step(a, bf, acc[nt]);
      }
    }
    const int rowb = mt * 16 + (hi ? 8 : 0);
#pragma unroll
    for (int nt = 0; nt < 4; ++nt) {
      v8h pk;
#pragma unroll
      for (int r = 0; r < 8; ++r)
        pk[r] = (_Float16)(fmaxf(acc[nt][r] + Ub[rowb + r], 0.f) *
                           wvec[b * C_SX2 + rowb + r]);
      *(v8h*)(&us[(nt * 16 + lr) * P256 + rowb]) = pk;
    }
  }
  __syncthreads();

  // V (128x256): 8 M-tiles / 8 waves
  {
    const int mt = wave;
    v8f acc[4] = {};
    for (int ks = 0; ks < C_SX2 / 32; ++ks) {
      const int k0 = ks * 32;
      v16h a = load_a_f16(Vw, C_SX2, mt * 16, k0, lane);
#pragma unroll
      for (int nt = 0; nt < 4; ++nt) {
        v16h bf = load_b_ldsT(&us[(nt * 16 + lr) * P256], k0, lane);
        acc[nt] = wmma_step(a, bf, acc[nt]);
      }
    }
    const int rowb = mt * 16 + (hi ? 8 : 0);
#pragma unroll
    for (int nt = 0; nt < 4; ++nt) {
      v8h pk;
#pragma unroll
      for (int r = 0; r < 8; ++r)
        pk[r] = (_Float16)fmaxf(acc[nt][r] + Vb[rowb + r], 0.f);
      *(v8h*)(&bcp[(size_t)(p0 + nt * 16 + lr) * CB + rowb]) = pk;
    }
  }

  // append tx channels 128..191 (pixel-major, b128 stores)
  {
    const int p = threadIdx.x & 63;
    const int ccg = threadIdx.x >> 6;  // 0..3, 16 channels each
    float v[16];
#pragma unroll
    for (int j = 0; j < 16; ++j)
      v[j] = tx[((size_t)b * C_T + ccg * 16 + j) * HW + p0 + p];
    v8h pk0, pk1;
#pragma unroll
    for (int j = 0; j < 8; ++j) {
      pk0[j] = (_Float16)v[j];
      pk1[j] = (_Float16)v[8 + j];
    }
    _Float16* dst = &bcp[(size_t)(p0 + p) * CB + EBD + ccg * 16];
    *(v8h*)(dst)     = pk0;
    *(v8h*)(dst + 8) = pk1;
  }
}

// ---------------------------------------------------------------------------
// K4: async-stage bcat tile, a1-GEMM -> ReLU -> a2-GEMM -> sigmoid -> gate
//     -> gact (pixel-major f16)
// ---------------------------------------------------------------------------
__global__ __launch_bounds__(256) void k4_attn(
    const _Float16* __restrict__ a1w, const float* __restrict__ a1b,
    const _Float16* __restrict__ a2w, const float* __restrict__ a2b,
    const _Float16* __restrict__ bcat, _Float16* __restrict__ gact) {
  __shared__ _Float16 bs[64 * P192];   // [col][c] 25 KB
  __shared__ _Float16 as1[64 * P128];  // [col][c] 17 KB
  const int b    = blockIdx.y;
  const int p0   = blockIdx.x * 64;
  const int wave = threadIdx.x >> 5;
  const int lane = threadIdx.x & 31;
  const int lr   = lane & 15;
  const bool hi  = (lane & 16) != 0;
  const _Float16* bcp = bcat + (size_t)b * HW * CB;
  _Float16* gcp       = gact + (size_t)b * HW * CB;

  // 64 px x 192 ch, 16B chunks via async global->LDS
  for (int q = threadIdx.x; q < 64 * (CB / 8); q += 256) {
    const int p = q / (CB / 8), c8 = q % (CB / 8);
    async_load_b128(lds_off(&bs[p * P192 + c8 * 8]),
                    bcp + (size_t)(p0 + p) * CB + c8 * 8);
  }
  wait_async0();
  __syncthreads();

  // GEMM1: a1 = relu(A1[128x192] @ bcat + b1)
  {
    const int mt = wave;
    v8f acc[4] = {};
    for (int ks = 0; ks < CB / 32; ++ks) {
      const int k0 = ks * 32;
      v16h a = load_a_f16(a1w, CB, mt * 16, k0, lane);
#pragma unroll
      for (int nt = 0; nt < 4; ++nt) {
        v16h bf = load_b_ldsT(&bs[(nt * 16 + lr) * P192], k0, lane);
        acc[nt] = wmma_step(a, bf, acc[nt]);
      }
    }
    const int rowb = mt * 16 + (hi ? 8 : 0);
#pragma unroll
    for (int nt = 0; nt < 4; ++nt) {
      v8h pk;
#pragma unroll
      for (int r = 0; r < 8; ++r)
        pk[r] = (_Float16)fmaxf(acc[nt][r] + a1b[rowb + r], 0.f);
      *(v8h*)(&as1[(nt * 16 + lr) * P128 + rowb]) = pk;
    }
  }
  __syncthreads();

  // GEMM2: a2 = sigmoid(A2[192x128] @ a1 + b2); gate; 12x4 tiles / 8 waves
  for (int j = 0; j < 6; ++j) {
    const int id = wave + 8 * j;
    const int mt = id >> 2, nt = id & 3;
    const int col = nt * 16 + lr;
    v8f c = {};
    for (int ks = 0; ks < A1C / 32; ++ks) {
      v16h a = load_a_f16(a2w, A1C, mt * 16, ks * 32, lane);
      v16h bf = load_b_ldsT(&as1[col * P128], ks * 32, lane);
      c = wmma_step(a, bf, c);
    }
    const int rowb = mt * 16 + (hi ? 8 : 0);
    v8h bv = *(const v8h*)(&bs[col * P192 + rowb]);
    v8h pk;
#pragma unroll
    for (int r = 0; r < 8; ++r) {
      const float s = 1.f / (1.f + expf(-(c[r] + a2b[rowb + r])));
      pk[r] = (_Float16)(s * (float)bv[r]);
    }
    *(v8h*)(&gcp[(size_t)(p0 + col) * CB + rowb]) = pk;
  }
}

// ---------------------------------------------------------------------------
// K5: 3x3 SAME conv (192->256). Block = 2 rows x 64 px; halo 4x66 positions
// async-staged into LDS [pos][c]; K = tap*192 + c; 8 N-tiles per A load.
// ---------------------------------------------------------------------------
__global__ __launch_bounds__(256) void k5_conv3(
    const _Float16* __restrict__ gact, const _Float16* __restrict__ otw,
    const float* __restrict__ otb, float* __restrict__ out) {
  __shared__ _Float16 ins[264 * P192];  // 4*66 positions x 192 ch, ~106 KB
  const int b  = blockIdx.z;
  const int y0 = blockIdx.y * 2;
  const int x0 = blockIdx.x * 64;
  const _Float16* gb = gact + (size_t)b * HW * CB;

  for (int q = threadIdx.x; q < 264 * (CB / 8); q += 256) {
    const int pos = q / (CB / 8), c8 = q % (CB / 8);
    const int dy4 = pos / 66, col = pos % 66;
    const int ys = y0 - 1 + dy4, xs = x0 - 1 + col;
    if (ys >= 0 && ys < H && xs >= 0 && xs < Wd) {
      async_load_b128(lds_off(&ins[pos * P192 + c8 * 8]),
                      gb + ((size_t)ys * Wd + xs) * CB + c8 * 8);
    } else {
      v8h z = {};
      *(v8h*)(&ins[pos * P192 + c8 * 8]) = z;
    }
  }
  wait_async0();
  __syncthreads();

  const int wave = threadIdx.x >> 5;
  const int lane = threadIdx.x & 31;
  const int lr   = lane & 15;
  const bool hi  = (lane & 16) != 0;

  for (int hf = 0; hf < 2; ++hf) {        // 16 M-tiles / 8 waves
    const int mt = wave + hf * 8;
    v8f acc[8] = {};
    for (int ks = 0; ks < KOT / 32; ++ks) {  // 54 K-steps
      const int k0  = ks * 32;
      const int tap = k0 / CB;             // constant within a k-step
      const int dy  = tap / 3, dx = tap % 3;
      const int kc  = k0 % CB;
      v16h a = load_a_f16(otw, KOT, mt * 16, k0, lane);
#pragma unroll
      for (int nt = 0; nt < 8; ++nt) {
        const int pl = nt * 16 + lr;       // 0..127
        const int yy = pl >> 6, xx = pl & 63;
        const int pos = (yy + dy) * 66 + xx + dx;
        v16h bf = load_b_ldsT(&ins[pos * P192], kc, lane);
        acc[nt] = wmma_step(a, bf, acc[nt]);
      }
    }
    const int rowb = mt * 16 + (hi ? 8 : 0);
#pragma unroll
    for (int nt = 0; nt < 8; ++nt) {
      const int pl = nt * 16 + lr;
      const int yy = pl >> 6, xx = pl & 63;
#pragma unroll
      for (int r = 0; r < 8; ++r) {
        out[((size_t)b * OTC + rowb + r) * HW + (y0 + yy) * Wd + x0 + xx] =
            fmaxf(acc[nt][r] + otb[rowb + r], 0.f);
      }
    }
  }
}

// ---------------------------------------------------------------------------
extern "C" void kernel_launch(void* const* d_in, const int* in_sizes, int n_in,
                              void* d_out, int out_size, void* d_ws, size_t ws_size,
                              hipStream_t stream) {
  const float* sx1 = (const float*)d_in[0];
  const float* sx2 = (const float*)d_in[1];
  const float* tx  = (const float*)d_in[2];
  const float* t1w = (const float*)d_in[3];
  const float* t1b = (const float*)d_in[4];
  const float* t2w = (const float*)d_in[5];
  const float* t3w = (const float*)d_in[6];
  const float* Uw  = (const float*)d_in[7];
  const float* Ub  = (const float*)d_in[8];
  const float* Vw  = (const float*)d_in[9];
  const float* Vb  = (const float*)d_in[10];
  const float* a1w = (const float*)d_in[11];
  const float* a1b = (const float*)d_in[12];
  const float* a2w = (const float*)d_in[13];
  const float* a2b = (const float*)d_in[14];
  const float* otw = (const float*)d_in[15];
  const float* otb = (const float*)d_in[16];

  char* ws = (char*)d_ws;
  float* gsum = (float*)ws;                          // 8*128 f32
  float* wvec = (float*)(ws + 4096);                 // 8*256 f32

  size_t off = 16384;
  _Float16* t1w16 = (_Float16*)(ws + off); off += (size_t)E1  * C1   * 2;
  _Float16* Uw16  = (_Float16*)(ws + off); off += (size_t)C_SX2*C_SX2* 2;
  _Float16* Vw16  = (_Float16*)(ws + off); off += (size_t)EBD * C_SX2* 2;
  _Float16* a1w16 = (_Float16*)(ws + off); off += (size_t)A1C * CB   * 2;
  _Float16* a2w16 = (_Float16*)(ws + off); off += (size_t)CB  * A1C  * 2;
  _Float16* otw16 = (_Float16*)(ws + off); off += (size_t)OTC * KOT  * 2;
  off = (off + 4095) & ~(size_t)4095;

  _Float16* bcat = (_Float16*)(ws + off); off += (size_t)Bn * HW * CB * 2;
  _Float16* gact = (_Float16*)(ws + off);

  float* out  = (float*)d_out;
  float* wout = out + (size_t)Bn * OTC * HW;

  auto cvt = [&](const float* s, _Float16* d, int n) {
    kcvt_f16<<<(n + 255) / 256, 256, 0, stream>>>(s, d, n);
  };
  cvt(t1w, t1w16, E1 * C1);
  cvt(Uw,  Uw16,  C_SX2 * C_SX2);
  cvt(Vw,  Vw16,  EBD * C_SX2);
  cvt(a1w, a1w16, A1C * CB);
  cvt(a2w, a2w16, CB * A1C);
  kperm_ot<<<(OTC * KOT + 255) / 256, 256, 0, stream>>>(otw, otw16);

  k0_zero<<<(Bn * E1 + 255) / 256, 256, 0, stream>>>(gsum, Bn * E1);
  k1_t1_gap<<<dim3(HW / 64, Bn), 256, 0, stream>>>(sx1, tx, t1w16, t1b, gsum);
  k2_transfer<<<Bn, 256, 0, stream>>>(gsum, t2w, t3w, wvec, wout);
  k3_mod<<<dim3(HW / 64, Bn), 256, 0, stream>>>(sx2, tx, Uw16, Ub, Vw16, Vb,
                                                wvec, bcat);
  k4_attn<<<dim3(HW / 64, Bn), 256, 0, stream>>>(a1w16, a1b, a2w16, a2b,
                                                 bcat, gact);
  k5_conv3<<<dim3(Wd / 64, H / 2, Bn), 256, 0, stream>>>(gact, otw16, otb, out);
}
